// Graph_Encoder_76888504533549
// MI455X (gfx1250) — matrix operand order repaired and checked by reading, not compiled, not run on hardware
//
#include <hip/hip_runtime.h>
#include <stdint.h>

typedef _Float16 half_t;
typedef __attribute__((ext_vector_type(16))) _Float16 v16h;
typedef __attribute__((ext_vector_type(8)))  float    v8f;
typedef __attribute__((ext_vector_type(4)))  unsigned v4u;
typedef __attribute__((ext_vector_type(8)))  unsigned v8u;

#define D 128
#define NEG_SLOPE 0.2f
#define EPS_F 1e-16f

// ---------- helpers ----------
__device__ __forceinline__ unsigned fenc(float f) {
    unsigned u = __float_as_uint(f);
    return (u & 0x80000000u) ? ~u : (u | 0x80000000u);
}
__device__ __forceinline__ float fdec(unsigned u) {
    unsigned b = (u & 0x80000000u) ? (u & 0x7FFFFFFFu) : ~u;
    return __uint_as_float(b);
}
__device__ __forceinline__ float lrelu(float v) { return v > 0.f ? v : NEG_SLOPE * v; }

__device__ __forceinline__ void wait_tensorcnt0() {
#if __has_builtin(__builtin_amdgcn_s_wait_tensorcnt)
    __builtin_amdgcn_s_wait_tensorcnt(0);
#else
    asm volatile("s_wait_tensorcnt 0x0" ::: "memory");
#endif
}

// ---------- kernels ----------
__global__ void k_f32_to_f16(const float* __restrict__ src, half_t* __restrict__ dst, int n) {
    int i = blockIdx.x * blockDim.x + threadIdx.x;
    if (i < n) dst[i] = (half_t)src[i];
}

// H[M,128] = X[M,128] @ Wh[128,128]   (WMMA f16 -> f32 accumulate)
// X tile is staged into LDS by the Tensor Data Mover (tensor_load_to_lds).
__global__ __launch_bounds__(256) void k_gemm_wmma(const float* __restrict__ X,
                                                   const half_t* __restrict__ Wh,
                                                   float* __restrict__ H, int M) {
    __shared__ float xs[16 * D];   // 8KB f32 tile at LDS offset 0
    const int tid  = threadIdx.x;
    const int row0 = blockIdx.x * 16;

    // ---- TDM: DMA the 16x128 f32 tile (rows clamped; OOB reads return 0) ----
    if (tid < 32) {                       // one wave issues; TDM ignores EXEC
        unsigned rows = (unsigned)(M - row0);
        if (rows > 16u) rows = 16u;
        uint64_t ga = (uint64_t)(const void*)(X + (size_t)row0 * D);
        v4u g0;
        g0[0] = 1u;                              // count=1, user descriptor
        g0[1] = 0u;                              // lds_addr = 0 (xs at LDS base)
        g0[2] = (unsigned)ga;                    // global_addr[31:0]
        g0[3] = (unsigned)(ga >> 32) | (2u << 30); // global_addr[56:32] | type=2
        v8u g1;
        g1[0] = 2u << 16;                        // data_size = 4 bytes
        g1[1] = (unsigned)D << 16;               // tensor_dim0 = 128  (bits 79:48)
        g1[2] = rows << 16;                      // tensor_dim1 = rows (bits 111:80)
        g1[3] = (unsigned)D << 16;               // tile_dim0 = 128    (bits 127:112)
        g1[4] = 16u;                             // tile_dim1 = 16     (bits 143:128)
        g1[5] = (unsigned)D;                     // tensor_dim0_stride = 128 (bits 207:160)
        g1[6] = 0u;
        g1[7] = 0u;
        asm volatile("tensor_load_to_lds %0, %1" :: "s"(g0), "s"(g1) : "memory");
        wait_tensorcnt0();
    }
    __syncthreads();

    const int wave = tid >> 5;          // 0..7 -> 16-col tile
    const int lane = tid & 31;
    const int col0 = wave * 16;
    const int m    = lane & 15;
    const int khi  = (lane >> 4) << 3;  // 0 or 8

    v8f acc = {};
#pragma unroll
    for (int kb = 0; kb < D; kb += 32) {
        v16h a;
#pragma unroll
        for (int e = 0; e < 16; ++e) {
            int k = kb + (e & 7) + ((e >> 3) << 4) + khi;   // ISA 16-bit A layout
            a[e] = (half_t)xs[m * D + k];
        }
        // B layout: lane%16 = N, element e = K (same order as A)
        v16h b;
#pragma unroll
        for (int e = 0; e < 16; ++e) {
            int k = kb + (e & 7) + ((e >> 3) << 4) + khi;
            b[e] = Wh[(size_t)k * D + col0 + m];
        }
        acc = __builtin_amdgcn_wmma_f32_16x16x32_f16(false, a, false, b,
                                                     (short)0, acc, false, false);
    }

    // C/D layout: lane -> col, vgpr v -> row = v + 8*(lane>>4)
    const int col   = col0 + (lane & 15);
    const int rbase = (lane >> 4) << 3;
    if (row0 + 16 <= M) {               // uniform fast path: full tile
#pragma unroll
        for (int v = 0; v < 8; ++v)
            H[(size_t)(row0 + rbase + v) * D + col] = acc[v];
    } else {
#pragma unroll
        for (int v = 0; v < 8; ++v) {
            int r = row0 + rbase + v;
            if (r < M) H[(size_t)r * D + col] = acc[v];
        }
    }
}

// per-node attention logits: a_s[n] = h[n].att_s ; a_d[n] = h[n].att_d
__global__ __launch_bounds__(256) void k_dots(const float4* __restrict__ H4,
                                              const float4* __restrict__ as4,
                                              const float4* __restrict__ ad4,
                                              float* __restrict__ a_s,
                                              float* __restrict__ a_d, int N) {
    int lane = threadIdx.x & 31;
    int node = blockIdx.x * (blockDim.x >> 5) + (threadIdx.x >> 5);
    if (node >= N) return;
    float4 h = H4[(size_t)node * 32 + lane];
    float4 s = as4[lane], d = ad4[lane];
    float ps = h.x * s.x + h.y * s.y + h.z * s.z + h.w * s.w;
    float pd = h.x * d.x + h.y * d.y + h.z * d.z + h.w * d.w;
#pragma unroll
    for (int o = 16; o; o >>= 1) {
        ps += __shfl_down(ps, o, 32);
        pd += __shfl_down(pd, o, 32);
    }
    if (lane == 0) { a_s[node] = ps; a_d[node] = pd; }
}

// pass 1: segment max (encoded-uint atomicMax)
__global__ void k_edge_max(const int* __restrict__ ei, const float* __restrict__ a_s,
                           const float* __restrict__ a_d, unsigned* __restrict__ m_enc,
                           int E, int N) {
    int e = blockIdx.x * blockDim.x + threadIdx.x;
    int Etot = E + N;
    if (e >= Etot) return;
    int s, d;
    if (e < E) { s = ei[e]; d = ei[E + e]; } else { s = d = e - E; }
    float v = lrelu(a_s[s] + a_d[d]);
    atomicMax(&m_enc[d], fenc(v));
}

// pass 2: segment sum of exp(v - m)
__global__ void k_edge_expsum(const int* __restrict__ ei, const float* __restrict__ a_s,
                              const float* __restrict__ a_d, const unsigned* __restrict__ m_enc,
                              float* __restrict__ denom, int E, int N) {
    int e = blockIdx.x * blockDim.x + threadIdx.x;
    int Etot = E + N;
    if (e >= Etot) return;
    int s, d;
    if (e < E) { s = ei[e]; d = ei[E + e]; } else { s = d = e - E; }
    float v = lrelu(a_s[s] + a_d[d]);
    atomicAdd(&denom[d], expf(v - fdec(m_enc[d])));
}

// pass 3: agg[dst] += alpha * h[src]   (one wave per edge, float4 per lane)
__global__ __launch_bounds__(256) void k_edge_agg(const int* __restrict__ ei,
                                                  const float* __restrict__ a_s,
                                                  const float* __restrict__ a_d,
                                                  const unsigned* __restrict__ m_enc,
                                                  const float* __restrict__ denom,
                                                  const float4* __restrict__ H4,
                                                  float* __restrict__ agg, int E, int N) {
    int lane = threadIdx.x & 31;
    int e = blockIdx.x * 8 + (threadIdx.x >> 5);
    int Etot = E + N;
    if (e >= Etot) return;
    int s, d;
    if (e < E) { s = ei[e]; d = ei[E + e]; } else { s = d = e - E; }
    float v = lrelu(a_s[s] + a_d[d]);
    float alpha = expf(v - fdec(m_enc[d])) / (denom[d] + EPS_F);
    float4 h = H4[(size_t)s * 32 + lane];
    float* o = agg + (size_t)d * D + lane * 4;
    atomicAdd(o + 0, alpha * h.x);
    atomicAdd(o + 1, alpha * h.y);
    atomicAdd(o + 2, alpha * h.z);
    atomicAdd(o + 3, alpha * h.w);
}

// in-place: x = relu(x + b)
__global__ void k_bias_relu(float* __restrict__ x, const float* __restrict__ b, int n) {
    int i = blockIdx.x * blockDim.x + threadIdx.x;
    if (i < n) {
        float v = x[i] + b[i & (D - 1)];
        x[i] = v > 0.f ? v : 0.f;
    }
}

// d_out[c] = N * b2[c]  (bias of layer 2 summed over the pool)
__global__ void k_init_out(float* __restrict__ out, const float* __restrict__ b, int N) {
    int c = threadIdx.x;
    if (c < D) out[c] = (float)N * b[c];
}

// d_out[c] += sum_n agg[n][c]
__global__ __launch_bounds__(128) void k_reduce(const float* __restrict__ agg,
                                                float* __restrict__ out, int N) {
    int c = threadIdx.x;   // 0..127
    float s = 0.f;
    for (int r = blockIdx.x; r < N; r += gridDim.x)
        s += agg[(size_t)r * D + c];
    atomicAdd(&out[c], s);
}

// ---------- host-side layer driver ----------
static void run_layer(const float* xin, const half_t* wh, const float* att_s,
                      const float* att_d, const int* ei, int E, int N,
                      float* h, float* agg, float* a_s, float* a_d,
                      unsigned* m_enc, float* denom, hipStream_t stream) {
    int Etot = E + N;
    k_gemm_wmma<<<(N + 15) / 16, 256, 0, stream>>>(xin, wh, h, N);
    k_dots<<<(N + 7) / 8, 256, 0, stream>>>((const float4*)h, (const float4*)att_s,
                                            (const float4*)att_d, a_s, a_d, N);
    // zero agg and (m_enc, denom) which are contiguous
    hipMemsetAsync(agg, 0, (size_t)N * D * sizeof(float), stream);
    hipMemsetAsync(m_enc, 0, (size_t)N * 2 * sizeof(unsigned), stream);
    k_edge_max<<<(Etot + 255) / 256, 256, 0, stream>>>(ei, a_s, a_d, m_enc, E, N);
    k_edge_expsum<<<(Etot + 255) / 256, 256, 0, stream>>>(ei, a_s, a_d, m_enc, denom, E, N);
    k_edge_agg<<<(Etot + 7) / 8, 256, 0, stream>>>(ei, a_s, a_d, m_enc, denom,
                                                   (const float4*)h, agg, E, N);
}

extern "C" void kernel_launch(void* const* d_in, const int* in_sizes, int n_in,
                              void* d_out, int out_size, void* d_ws, size_t ws_size,
                              hipStream_t stream) {
    const float* x   = (const float*)d_in[0];
    const int*   ei  = (const int*)  d_in[1];
    const float* W1  = (const float*)d_in[2];
    const float* as1 = (const float*)d_in[3];
    const float* ad1 = (const float*)d_in[4];
    const float* b1  = (const float*)d_in[5];
    const float* W2  = (const float*)d_in[6];
    const float* as2 = (const float*)d_in[7];
    const float* ad2 = (const float*)d_in[8];
    const float* b2  = (const float*)d_in[9];

    const int N = in_sizes[0] / D;
    const int E = in_sizes[1] / 2;

    // workspace layout (all offsets 32B-aligned)
    char* ws = (char*)d_ws;
    half_t*   wh1   = (half_t*)ws;                    ws += (size_t)D * D * sizeof(half_t);
    half_t*   wh2   = (half_t*)ws;                    ws += (size_t)D * D * sizeof(half_t);
    float*    h     = (float*)ws;                     ws += (size_t)N * D * sizeof(float);
    float*    agg   = (float*)ws;                     ws += (size_t)N * D * sizeof(float);
    float*    a_s   = (float*)ws;                     ws += (size_t)N * sizeof(float);
    float*    a_d   = (float*)ws;                     ws += (size_t)N * sizeof(float);
    unsigned* m_enc = (unsigned*)ws;                  ws += (size_t)N * sizeof(unsigned);
    float*    denom = (float*)ws;                     ws += (size_t)N * sizeof(float);

    // convert weights to f16 once per call
    k_f32_to_f16<<<(2 * D * D + 255) / 256, 256, 0, stream>>>(W1, wh1, D * D);
    k_f32_to_f16<<<(2 * D * D + 255) / 256, 256, 0, stream>>>(W2, wh2, D * D);

    // ---- layer 1 ----
    run_layer(x, wh1, as1, ad1, ei, E, N, h, agg, a_s, a_d, m_enc, denom, stream);
    // x2 = relu(agg + b1), in place (agg becomes layer-2 input)
    k_bias_relu<<<((size_t)N * D + 255) / 256, 256, 0, stream>>>(agg, b1, N * D);

    // ---- layer 2 ---- (gemm reads agg as input, writes h; agg re-zeroed only
    //                    after gemm+dots have consumed it)
    {
        int Etot = E + N;
        k_gemm_wmma<<<(N + 15) / 16, 256, 0, stream>>>(agg, wh2, h, N);
        k_dots<<<(N + 7) / 8, 256, 0, stream>>>((const float4*)h, (const float4*)as2,
                                                (const float4*)ad2, a_s, a_d, N);
        hipMemsetAsync(agg, 0, (size_t)N * D * sizeof(float), stream);
        hipMemsetAsync(m_enc, 0, (size_t)N * 2 * sizeof(unsigned), stream);
        k_edge_max<<<(Etot + 255) / 256, 256, 0, stream>>>(ei, a_s, a_d, m_enc, E, N);
        k_edge_expsum<<<(Etot + 255) / 256, 256, 0, stream>>>(ei, a_s, a_d, m_enc, denom, E, N);
        k_edge_agg<<<(Etot + 7) / 8, 256, 0, stream>>>(ei, a_s, a_d, m_enc, denom,
                                                       (const float4*)h, agg, E, N);
    }

    // ---- global add pool: out = N*b2 + sum_n agg[n] ----
    float* out = (float*)d_out;
    k_init_out<<<1, D, 0, stream>>>(out, b2, N);
    k_reduce<<<512, D, 0, stream>>>(agg, out, N);
}
